// OuterProduct_45191645888851
// MI455X (gfx1250) — matrix operand order repaired and checked by reading, not compiled, not run on hardware
//
#include <hip/hip_runtime.h>
#include <hip/hip_bf16.h>

// ---------------------------------------------------------------------------
// value[b] = -(x[b] . z[b])^2 with z = MLP(x): 64->512->512->256->64, ELU.
// Fully fused; each wave carries 16 rows through all 4 layers with
// v_wmma_f32_16x16x32_f16. Weights pre-packed to f16 WMMA-B tile blocks in ws.
// 4 independent accumulator chains per group for XDL ILP; A-fragments reused
// 4x per LDS read; weight loads fully coalesced (lane*32B within 1KB blocks).
// ---------------------------------------------------------------------------

typedef __attribute__((ext_vector_type(16))) _Float16 v16h;
typedef __attribute__((ext_vector_type(8)))  _Float16 v8h;
typedef __attribute__((ext_vector_type(4)))  _Float16 v4h;
typedef __attribute__((ext_vector_type(8)))  float    v8f;

#define BATCH   65536
#define MBLOCK  128          // rows per workgroup (8 waves x 16 rows)
#define SX      72           // LDS row stride (halves): 64 + 8 pad
#define SA      520          // LDS row stride (halves): 512 + 8 pad

// f16 packed-weight layout inside d_ws (element offsets)
#define W1_OFF  0            // 512x64
#define W2_OFF  32768        // 512x512
#define W3_OFF  294912       // 256x512
#define W4_OFF  425984       // 64x256
#define WTOT    442368

#define SMEM_HALVES (MBLOCK*SX + 2*MBLOCK*SA)          // 142336 halves
#define SMEM_BYTES  (SMEM_HALVES * 2)                  // 284672 B (< 320KB WGP LDS)

// ---------------- weight f32 -> f16 pack prologue --------------------------
// Destination layout: blocks of 512 halves, one per (nt, ks):
//   dst[((nt*KS + ks)*32 + lane)*16 + j] = W[nt*16 + (lane&15)][ks*32 + (j>>3)*16 + (lane>>4)*8 + (j&7)]
// so in the GEMM every lane loads 32 contiguous bytes at base + lane*32.
__global__ void pack_weights(const float* __restrict__ W1, const float* __restrict__ W2,
                             const float* __restrict__ W3, const float* __restrict__ W4,
                             _Float16* __restrict__ o) {
    int i = blockIdx.x * blockDim.x + threadIdx.x;
    if (i >= WTOT) return;
    const float* src; int K; int il;
    if (i < W2_OFF)      { src = W1; K = 64;  il = i - W1_OFF; }
    else if (i < W3_OFF) { src = W2; K = 512; il = i - W2_OFF; }
    else if (i < W4_OFF) { src = W3; K = 512; il = i - W3_OFF; }
    else                 { src = W4; K = 256; il = i - W4_OFF; }
    int j    = il & 15;
    int lane = (il >> 4) & 31;
    int blk  = il >> 9;          // nt*KS + ks
    int KS   = K >> 5;
    int nt   = blk / KS;
    int ks   = blk - nt * KS;
    int n    = nt * 16 + (lane & 15);
    int k    = ks * 32 + (j >> 3) * 16 + (lane >> 4) * 8 + (j & 7);
    o[i] = (_Float16)src[n * K + k];
}

// ---------------- one MLP layer for a wave's 16-row slab -------------------
// ldsIn : wave-local input slab [16 x K] (row stride SIN halves)
// ldsOut: wave-local output slab [16 x N] (row stride SOUT halves)
// Wg    : packed f16 weights for this layer (tile-block layout above)
template<int K, int N, int SIN, int SOUT, bool DO_ACT>
__device__ __forceinline__ void mlp_layer(const _Float16* __restrict__ ldsIn,
                                          _Float16* __restrict__ ldsOut,
                                          const _Float16* __restrict__ Wg,
                                          const float* __restrict__ bias,
                                          int lane) {
    const int lo = lane & 15;
    const int hi = lane >> 4;
    constexpr int KS = K / 32;
    // A layout (16x32 f16): lane holds row M=lo; chunks at K = kb+8*hi, kb+16+8*hi
    const _Float16* arow = ldsIn + lo * SIN + hi * 8;
    const _Float16* wl   = Wg + lane * 16;
#pragma unroll 1
    for (int ng = 0; ng < N / 64; ++ng) {            // 4 N-tiles per group
        v8f acc[4];
#pragma unroll
        for (int t = 0; t < 4; ++t) acc[t] = (v8f){0,0,0,0,0,0,0,0};
#pragma unroll 2
        for (int ks = 0; ks < KS; ++ks) {
            const int kb = ks * 32;
            v8h a0 = *(const v8h*)(arow + kb);
            v8h a1 = *(const v8h*)(arow + kb + 16);
            v16h A = __builtin_shufflevector(a0, a1, 0,1,2,3,4,5,6,7,8,9,10,11,12,13,14,15);
#pragma unroll
            for (int t = 0; t < 4; ++t) {            // 4 independent WMMA chains
                const _Float16* wb = wl + ((size_t)((ng * 4 + t) * KS + ks) << 9);
                v8h b0 = *(const v8h*)(wb);
                v8h b1 = *(const v8h*)(wb + 8);
                v16h B = __builtin_shufflevector(b0, b1, 0,1,2,3,4,5,6,7,8,9,10,11,12,13,14,15);
                acc[t] = __builtin_amdgcn_wmma_f32_16x16x32_f16(false, A, false, B,
                                                                (short)0, acc[t], false, false);
            }
        }
#pragma unroll
        for (int t = 0; t < 4; ++t) {
            const int nt = ng * 4 + t;
            float bv = bias[nt * 16 + lo];
#pragma unroll
            for (int r = 0; r < 8; ++r) {
                // D layout: vgpr r holds row M = r + 8*hi, column N = lo
                float v = acc[t][r] + bv;
                if (DO_ACT) v = (v > 0.0f) ? v : (__expf(v) - 1.0f);
                ldsOut[(size_t)(r + hi * 8) * SOUT + nt * 16 + lo] = (_Float16)v;
            }
        }
    }
}

// ---------------- fused MLP + quadratic-form kernel ------------------------
__global__ __launch_bounds__(256, 1)
void fused_mlp_quad(const float* __restrict__ x,
                    const _Float16* __restrict__ Wh,
                    const float* __restrict__ b1, const float* __restrict__ b2,
                    const float* __restrict__ b3, const float* __restrict__ b4,
                    float* __restrict__ out) {
    extern __shared__ _Float16 smem[];
    const int tid  = threadIdx.x;
    const int wid  = tid >> 5;
    const int lane = tid & 31;
    const int lo   = lane & 15;
    const int hi   = lane >> 4;

    _Float16* Xs = smem + (size_t)wid * 16 * SX;
    _Float16* As = smem + MBLOCK * SX + (size_t)wid * 16 * SA;
    _Float16* Bs = smem + MBLOCK * SX + MBLOCK * SA + (size_t)wid * 16 * SA;

    const int rowBase = blockIdx.x * MBLOCK + wid * 16;

    // ---- stage this wave's 16x64 x-tile (f32 -> f16) into Xs --------------
    const float4* xg = (const float4*)(x + (size_t)rowBase * 64);
#pragma unroll
    for (int it = 0; it < 8; ++it) {
        int idx = lane + it * 32;            // 256 float4 = 16 rows x 16 quads
        float4 v = xg[idx];
        int row  = idx >> 4;
        int col  = (idx & 15) * 4;
        v4h h = { (_Float16)v.x, (_Float16)v.y, (_Float16)v.z, (_Float16)v.w };
        *(v4h*)(Xs + (size_t)row * SX + col) = h;
    }
    // Waves only touch their own slabs -> zero barriers in the whole kernel.

    // ---- layers 1..3 (ELU) -----------------------------------------------
    mlp_layer< 64, 512, SX, SA, true >(Xs, As, Wh + W1_OFF, b1, lane);
    mlp_layer<512, 512, SA, SA, true >(As, Bs, Wh + W2_OFF, b2, lane);
    mlp_layer<512, 256, SA, SA, true >(Bs, As, Wh + W3_OFF, b3, lane);

    // ---- layer 4 (linear, N=64: one group of 4 tiles) fused with x.z ------
    float dot[8];
#pragma unroll
    for (int r = 0; r < 8; ++r) dot[r] = 0.0f;

    {
        constexpr int KS = 256 / 32;
        const _Float16* arow = As + lo * SA + hi * 8;
        const _Float16* wl   = Wh + W4_OFF + lane * 16;
        v8f acc[4];
#pragma unroll
        for (int t = 0; t < 4; ++t) acc[t] = (v8f){0,0,0,0,0,0,0,0};
#pragma unroll 2
        for (int ks = 0; ks < KS; ++ks) {
            const int kb = ks * 32;
            v8h a0 = *(const v8h*)(arow + kb);
            v8h a1 = *(const v8h*)(arow + kb + 16);
            v16h A = __builtin_shufflevector(a0, a1, 0,1,2,3,4,5,6,7,8,9,10,11,12,13,14,15);
#pragma unroll
            for (int t = 0; t < 4; ++t) {
                const _Float16* wb = wl + ((size_t)(t * KS + ks) << 9);
                v8h b0 = *(const v8h*)(wb);
                v8h b1 = *(const v8h*)(wb + 8);
                v16h B = __builtin_shufflevector(b0, b1, 0,1,2,3,4,5,6,7,8,9,10,11,12,13,14,15);
                acc[t] = __builtin_amdgcn_wmma_f32_16x16x32_f16(false, A, false, B,
                                                                (short)0, acc[t], false, false);
            }
        }
#pragma unroll
        for (int t = 0; t < 4; ++t) {
            float bv = b4[t * 16 + lo];
#pragma unroll
            for (int r = 0; r < 8; ++r) {
                float z  = acc[t][r] + bv;                                  // z[row][n]
                float xv = (float)Xs[(size_t)(r + hi * 8) * SX + t * 16 + lo];
                dot[r] += z * xv;                                           // partial over n
            }
        }
    }

    // reduce over the 16 lanes holding the columns of each row half
#pragma unroll
    for (int r = 0; r < 8; ++r) {
#pragma unroll
        for (int m = 1; m < 16; m <<= 1)
            dot[r] += __shfl_xor(dot[r], m, 32);
    }
    if (lo == 0) {
#pragma unroll
        for (int r = 0; r < 8; ++r) {
            float d = dot[r];
            out[rowBase + hi * 8 + r] = -d * d;     // sign = -1
        }
    }
}

// ---------------------------------------------------------------------------
extern "C" void kernel_launch(void* const* d_in, const int* in_sizes, int n_in,
                              void* d_out, int out_size, void* d_ws, size_t ws_size,
                              hipStream_t stream) {
    const float* x  = (const float*)d_in[0];
    const float* W1 = (const float*)d_in[1];
    const float* b1 = (const float*)d_in[2];
    const float* W2 = (const float*)d_in[3];
    const float* b2 = (const float*)d_in[4];
    const float* W3 = (const float*)d_in[5];
    const float* b3 = (const float*)d_in[6];
    const float* W4 = (const float*)d_in[7];
    const float* b4 = (const float*)d_in[8];
    float* out = (float*)d_out;
    _Float16* Wh = (_Float16*)d_ws;                 // 442368 halves = 884,736 B

    (void)in_sizes; (void)n_in; (void)out_size; (void)ws_size;

    static_assert(SMEM_BYTES <= 320 * 1024, "LDS over WGP budget");
    hipFuncSetAttribute((const void*)fused_mlp_quad,
                        hipFuncAttributeMaxDynamicSharedMemorySize, SMEM_BYTES);

    pack_weights<<<WTOT / 256, 256, 0, stream>>>(W1, W2, W3, W4, Wh);
    fused_mlp_quad<<<BATCH / MBLOCK, 256, SMEM_BYTES, stream>>>(x, Wh, b1, b2, b3, b4, out);
}
